// SparseGatedRNNCell_41437844472172
// MI455X (gfx1250) — matrix operand order repaired and checked by reading, not compiled, not run on hardware
//
#include <hip/hip_runtime.h>
#include <hip/hip_bf16.h>

// ---------------------------------------------------------------------------
// SparseGatedRNNCell on gfx1250 (MI455X): bf16 WMMA GEMMs with async
// global->LDS double buffering + fused softmax / layernorm epilogues.
// B=4096, I=512, H=1024, S=4096.
// ---------------------------------------------------------------------------

typedef __attribute__((ext_vector_type(16))) __bf16 v16bf;
typedef __attribute__((ext_vector_type(8)))  float  v8f;
typedef __attribute__((ext_vector_type(4)))  unsigned int v4u;

union Frag16 {
    v4u   q[2];
    v16bf v;
};

__device__ __forceinline__ unsigned short f2bf(float f) {
    // round-to-nearest-even fp32 -> bf16 (bit pattern)
    unsigned int u = __float_as_uint(f);
    unsigned int lsb = (u >> 16) & 1u;
    u += 0x7fffu + lsb;
    return (unsigned short)(u >> 16);
}

// Async DMA: 16 B per lane, memory -> LDS, tracked by ASYNCcnt.
__device__ __forceinline__ void async_b128(const unsigned short* gptr,
                                           unsigned short* sptr) {
    unsigned lds = (unsigned)(uintptr_t)sptr;            // low 32 bits = LDS addr
    unsigned long long ga = (unsigned long long)(uintptr_t)gptr;
    asm volatile("global_load_async_to_lds_b128 %0, %1, off"
                 :: "v"(lds), "v"(ga) : "memory");
}

// ------------------------- packing kernels ---------------------------------

// A_act[b, 0:512] = x[b,:], A_act[b, 512:1536] = hx[b,:]  (cols 1536.. by softmax)
__global__ __launch_bounds__(256) void pack_act_kernel(
    const float* __restrict__ x, const float* __restrict__ hx,
    unsigned short* __restrict__ act) {
    int idx = blockIdx.x * 256 + threadIdx.x;          // over 4096*1536
    int b = idx / 1536, c = idx - b * 1536;
    float v = (c < 512) ? x[(size_t)b * 512 + c] : hx[(size_t)b * 1024 + (c - 512)];
    act[(size_t)b * 5632 + c] = f2bf(v);
}

// Wg[s, 0:512] = W_is[s,:], Wg[s, 512:1536] = W_hs[s,:]
__global__ __launch_bounds__(256) void pack_w1_kernel(
    const float* __restrict__ W_is, const float* __restrict__ W_hs,
    unsigned short* __restrict__ Wg) {
    int idx = blockIdx.x * 256 + threadIdx.x;          // over 4096*1536
    int s = idx / 1536, c = idx - s * 1536;
    float v = (c < 512) ? W_is[(size_t)s * 512 + c] : W_hs[(size_t)s * 1024 + (c - 512)];
    Wg[(size_t)s * 1536 + c] = f2bf(v);
}

// Wp[h, 0:512]=W_ih, [512:1536]=W_hh, [1536:5632]=W_sh
__global__ __launch_bounds__(256) void pack_w2_kernel(
    const float* __restrict__ W_ih, const float* __restrict__ W_hh,
    const float* __restrict__ W_sh, unsigned short* __restrict__ Wp) {
    int idx = blockIdx.x * 256 + threadIdx.x;          // over 1024*5632
    int h = idx / 5632, c = idx - h * 5632;
    float v;
    if (c < 512)        v = W_ih[(size_t)h * 512 + c];
    else if (c < 1536)  v = W_hh[(size_t)h * 1024 + (c - 512)];
    else                v = W_sh[(size_t)h * 4096 + (c - 1536)];
    Wp[(size_t)h * 5632 + c] = f2bf(v);
}

// ------------------------- bf16 WMMA GEMM ----------------------------------
// C[M,N] = A[M,K] * B[N,K]^T  (bf16 row-major operands, fp32 C).
// Tile: 128x128x32, 256 threads (8 waves, 2x4 wave grid, 64x32 per wave).
// Double-buffered LDS filled by GLOBAL_LOAD_ASYNC_TO_LDS_B128 (ASYNCcnt).

#define BM 128
#define BN 128
#define BKT 32
#define LDT 40   // LDS row pitch (ushorts) = 80 B: 20-bank skew, conflict-free rows

// one K-tile of WMMAs: 4x2 16x16 tiles per wave
__device__ __forceinline__ void compute_tile(
    const unsigned short* __restrict__ as, const unsigned short* __restrict__ bs,
    int tm0, int tn0, int l15, int half, v8f (&acc)[4][2]) {
    Frag16 bfrag[2], afrag[4];
    // B 32x16 bf16: lane = N (l15); per lane-half 16 contiguous K values
#pragma unroll
    for (int ni = 0; ni < 2; ++ni) {
        const unsigned short* p = &bs[(tn0 + ni * 16 + l15) * LDT + half * 16];
        bfrag[ni].q[0] = *reinterpret_cast<const v4u*>(p);
        bfrag[ni].q[1] = *reinterpret_cast<const v4u*>(p + 8);
    }
    // A 16x32 bf16: lane = M (l15); per lane-half 16 contiguous K values
#pragma unroll
    for (int mi = 0; mi < 4; ++mi) {
        const unsigned short* p = &as[(tm0 + mi * 16 + l15) * LDT + half * 16];
        afrag[mi].q[0] = *reinterpret_cast<const v4u*>(p);
        afrag[mi].q[1] = *reinterpret_cast<const v4u*>(p + 8);
    }
#pragma unroll
    for (int mi = 0; mi < 4; ++mi)
#pragma unroll
        for (int ni = 0; ni < 2; ++ni)
            acc[mi][ni] = __builtin_amdgcn_wmma_f32_16x16x32_bf16(
                false, afrag[mi].v, false, bfrag[ni].v,
                (short)0, acc[mi][ni], false, false);
}

__global__ __launch_bounds__(256) void gemm_bf16_wmma(
    const unsigned short* __restrict__ A, const unsigned short* __restrict__ Bw,
    float* __restrict__ C, int K, int lda, int ldb, int ldc) {
    __shared__ __align__(16) unsigned short As[2][BM * LDT];
    __shared__ __align__(16) unsigned short Bs[2][BN * LDT];

    const int m0 = blockIdx.x * BM;
    const int n0 = blockIdx.y * BN;
    const unsigned tid  = threadIdx.x;
    const unsigned lane = tid & 31u;
    const unsigned wid  = tid >> 5;
    const int half = (int)(lane >> 4);      // 0/1
    const int l15  = (int)(lane & 15u);
    const int tm0 = (int)(wid >> 2) * 64;   // wave row offset in tile
    const int tn0 = (int)(wid & 3u) * 32;   // wave col offset in tile

    v8f acc[4][2];
#pragma unroll
    for (int mi = 0; mi < 4; ++mi)
#pragma unroll
        for (int ni = 0; ni < 2; ++ni) acc[mi][ni] = (v8f){};

    // loader: 256 threads x 32 B -> 128x32 bf16 tile (2 threads per row)
    const int lrow = (int)(tid >> 1);        // 0..127
    const int lcol = (int)(tid & 1u) * 16;   // 0 or 16 (ushorts) = 0/32 B

    const unsigned short* gA = A + (size_t)(m0 + lrow) * lda + lcol;
    const unsigned short* gB = Bw + (size_t)(n0 + lrow) * ldb + lcol;
    const int sOff = lrow * LDT + lcol;
    unsigned short* sA[2] = { &As[0][sOff], &As[1][sOff] };
    unsigned short* sB[2] = { &Bs[0][sOff], &Bs[1][sOff] };

    // prologue: tile 0 -> buffer 0 (4 async ops per wave)
    async_b128(gA,     sA[0]);
    async_b128(gA + 8, sA[0] + 8);
    async_b128(gB,     sB[0]);
    async_b128(gB + 8, sB[0] + 8);
    gA += BKT; gB += BKT;

    const int T = K / BKT;   // >= 2 for all call sites
    int buf = 0;
    for (int i = 0; i < T - 1; ++i) {
        // unconditional prefetch of tile i+1 into the other buffer
        const int nb = buf ^ 1;
        async_b128(gA,     sA[nb]);
        async_b128(gA + 8, sA[nb] + 8);
        async_b128(gB,     sB[nb]);
        async_b128(gB + 8, sB[nb] + 8);
        gA += BKT; gB += BKT;
        asm volatile("s_wait_asynccnt 0x4" ::: "memory");  // tile i landed
        __syncthreads();   // every wave's slice of tile i visible
        compute_tile(As[buf], Bs[buf], tm0, tn0, l15, half, acc);
        __syncthreads();   // reads of buf done before it is refilled next iter
        buf = nb;
    }
    // peeled last tile
    asm volatile("s_wait_asynccnt 0x0" ::: "memory");
    __syncthreads();
    compute_tile(As[buf], Bs[buf], tm0, tn0, l15, half, acc);

    // D 16x16 f32: VGPR r -> m = r + 8*half, n = l15
#pragma unroll
    for (int mi = 0; mi < 4; ++mi)
#pragma unroll
        for (int ni = 0; ni < 2; ++ni)
#pragma unroll
            for (int r = 0; r < 8; ++r) {
                int m = m0 + tm0 + mi * 16 + half * 8 + r;
                int n = n0 + tn0 + ni * 16 + l15;
                C[(size_t)m * ldc + n] = acc[mi][ni][r];
            }
}

// ------------------------- softmax (row-wise, S=4096) ----------------------
// In-place on d_out sy region; also writes bf16 copy into A_act[:,1536:].
__global__ __launch_bounds__(256) void softmax_kernel(
    float* __restrict__ Sy, unsigned short* __restrict__ act) {
    __shared__ float red[256];
    const int b = blockIdx.x, tid = threadIdx.x;
    float* row = Sy + (size_t)b * 5120;       // base already offset by +1024 (col H)
    float v[16];
    float mx = -3.4e38f;
#pragma unroll
    for (int j = 0; j < 16; ++j) { v[j] = row[tid + j * 256]; mx = fmaxf(mx, v[j]); }
    red[tid] = mx; __syncthreads();
    for (int s = 128; s > 0; s >>= 1) {
        if (tid < s) red[tid] = fmaxf(red[tid], red[tid + s]);
        __syncthreads();
    }
    mx = red[0]; __syncthreads();
    float sum = 0.0f;
#pragma unroll
    for (int j = 0; j < 16; ++j) { v[j] = __expf(v[j] - mx); sum += v[j]; }
    red[tid] = sum; __syncthreads();
    for (int s = 128; s > 0; s >>= 1) {
        if (tid < s) red[tid] += red[tid + s];
        __syncthreads();
    }
    const float inv = 1.0f / red[0];
    unsigned short* arow = act + (size_t)b * 5632 + 1536;
#pragma unroll
    for (int j = 0; j < 16; ++j) {
        int idx = tid + j * 256;
        float o = v[j] * inv;
        row[idx] = o;
        arow[idx] = f2bf(o);
    }
}

// ------------------- layernorm (biased std) + relu, H=1024 -----------------
__global__ __launch_bounds__(256) void ln_relu_kernel(
    const float* __restrict__ pre, const float* __restrict__ internal,
    const float* __restrict__ mu, float* __restrict__ out) {
    __shared__ float r1[256], r2[256];
    const int b = blockIdx.x, tid = threadIdx.x;
    const float* prow = pre + (size_t)b * 1024;
    float v[4], s1 = 0.0f, s2 = 0.0f;
#pragma unroll
    for (int j = 0; j < 4; ++j) {
        v[j] = prow[tid + j * 256];
        s1 += v[j]; s2 += v[j] * v[j];
    }
    r1[tid] = s1; r2[tid] = s2; __syncthreads();
    for (int s = 128; s > 0; s >>= 1) {
        if (tid < s) { r1[tid] += r1[tid + s]; r2[tid] += r2[tid + s]; }
        __syncthreads();
    }
    const float m   = r1[0] * (1.0f / 1024.0f);
    float var = r2[0] * (1.0f / 1024.0f) - m * m;
    var = fmaxf(var, 0.0f);
    const float inv = 1.0f / (sqrtf(var) + 1e-4f);   // sd = sqrt(var) + EPS, SIG=1
    float* o1 = out + (size_t)b * 5120;                       // hy in concat
    float* o2 = out + (size_t)4096 * 5120 + (size_t)b * 1024; // second output hy
    const float* irow = internal + (size_t)b * 1024;
#pragma unroll
    for (int j = 0; j < 4; ++j) {
        int h = tid + j * 256;
        float ln = (v[j] - m) * inv + mu[h];
        float hy = fmaxf(ln + irow[h], 0.0f);
        o1[h] = hy;
        o2[h] = hy;
    }
}

// ---------------------------------------------------------------------------

extern "C" void kernel_launch(void* const* d_in, const int* in_sizes, int n_in,
                              void* d_out, int out_size, void* d_ws, size_t ws_size,
                              hipStream_t stream) {
    (void)in_sizes; (void)n_in; (void)out_size; (void)ws_size;

    const float* x        = (const float*)d_in[0];  // [4096,512]
    const float* internal = (const float*)d_in[1];  // [4096,1024]
    const float* hx       = (const float*)d_in[2];  // [4096,1024]
    const float* W_ih     = (const float*)d_in[3];  // [1024,512]
    const float* W_is     = (const float*)d_in[4];  // [4096,512]
    const float* W_sh     = (const float*)d_in[5];  // [1024,4096]
    const float* W_hh     = (const float*)d_in[6];  // [1024,1024]
    const float* W_hs     = (const float*)d_in[7];  // [4096,1024]
    const float* mu       = (const float*)d_in[8];  // [1024]
    float* out = (float*)d_out;                     // [4096,5120] ++ [4096,1024]

    // workspace layout (all 16B-aligned)
    char* ws = (char*)d_ws;
    unsigned short* A_act = (unsigned short*)(ws);            // 4096*5632 bf16 = 46,137,344 B
    unsigned short* Wg    = (unsigned short*)(ws + 46137344); // 4096*1536 bf16 = 12,582,912 B
    unsigned short* Wp    = (unsigned short*)(ws + 58720256); // 1024*5632 bf16 = 11,534,336 B
    float*          preb  = (float*)(ws + 70254592);          // 4096*1024 f32  = 16,777,216 B

    // 1) pack bf16 operands
    pack_act_kernel<<<(4096 * 1536) / 256, 256, 0, stream>>>(x, hx, A_act);
    pack_w1_kernel <<<(4096 * 1536) / 256, 256, 0, stream>>>(W_is, W_hs, Wg);
    pack_w2_kernel <<<(1024 * 5632) / 256, 256, 0, stream>>>(W_ih, W_hh, W_sh, Wp);

    // 2) s_pre = [x|hx] @ [W_is|W_hs]^T  -> directly into d_out sy region
    //    C base = out + col offset H (1024), ldc = H+S (5120)
    gemm_bf16_wmma<<<dim3(4096 / BM, 4096 / BN), 256, 0, stream>>>(
        A_act, Wg, out + 1024, /*K=*/1536, /*lda=*/5632, /*ldb=*/1536, /*ldc=*/5120);

    // 3) softmax rows (in place) + bf16 copy into A_act[:,1536:]
    softmax_kernel<<<4096, 256, 0, stream>>>(out + 1024, A_act);

    // 4) pre = [x|hx|sy] @ [W_ih|W_hh|W_sh]^T
    gemm_bf16_wmma<<<dim3(4096 / BM, 1024 / BN), 256, 0, stream>>>(
        A_act, Wp, preb, /*K=*/5632, /*lda=*/5632, /*ldb=*/5632, /*ldc=*/1024);

    // 5) layernorm + relu -> hy into both output regions
    ln_relu_kernel<<<4096, 256, 0, stream>>>(preb, internal, mu, out);
}
